// Network_26809185861895
// MI455X (gfx1250) — compile-verified, matrix operand order checked by reading, not run on hardware
//
#include <hip/hip_runtime.h>
#include <stdint.h>

typedef _Float16 half_t;
typedef __attribute__((ext_vector_type(16))) _Float16 v16h;
typedef __attribute__((ext_vector_type(8)))  _Float16 v8h;
typedef __attribute__((ext_vector_type(8)))  float    v8f;

#define S_DIM 16
#define B_DIM 64
#define N_AG  16
#define HIDN  256
#define NSAMP 16384   // S*B*N
#define NROWS 1024    // B*N
#define LATP  288     // 261 padded to 32-multiple

// ---------------------------------------------------------------------------
// WMMA fragment helpers (16-bit A/B layout per CDNA5 ISA 7.12.2):
// lanes 0-15 hold row/col (lane&15) with K = {0..7, 16..23} of the 32-chunk,
// lanes 16-31 hold K = {8..15, 24..31}.  Two b128 loads per fragment.
// ---------------------------------------------------------------------------
__device__ __forceinline__ v16h frag_ld(const half_t* p) {
  v8h lo = *(const v8h*)(p);
  v8h hi = *(const v8h*)(p + 16);
  return __builtin_shufflevector(lo, hi, 0,1,2,3,4,5,6,7,8,9,10,11,12,13,14,15);
}
__device__ __forceinline__ float wave_sum(float s) {
#pragma unroll
  for (int off = 16; off; off >>= 1) s += __shfl_xor(s, off, 32);
  return s;
}
__device__ __forceinline__ float wave_max(float s) {
#pragma unroll
  for (int off = 16; off; off >>= 1) s = fmaxf(s, __shfl_xor(s, off, 32));
  return s;
}
__device__ __forceinline__ float sigmoidf(float x) { return 1.0f / (1.0f + __expf(-x)); }

// ---------------------------------------------------------------------------
// Register-blocked, software-pipelined direct-WMMA GEMM, templated on the
// (shared) row stride LD == K so all intra-tile addresses are immediate
// offsets folded into the global_load_b128 IOFFSET field.  Only two base
// address registers stay live -> no spills around the pipelined loop.
// 32x64 output tile per wave (2 M-frags x 4 N-frags, 8 v8f accumulators);
// next chunk's 12 b128 loads are in flight while the current 8 WMMAs run.
// M mult of 32, N mult of 64.
// ---------------------------------------------------------------------------
template<int LD>
__global__ void gemm_f16_t(const half_t* __restrict__ A,
                           const half_t* __restrict__ Bw,
                           float* __restrict__ C, half_t* __restrict__ Ch, int ldc,
                           const float* __restrict__ bias, int nb,
                           int mblks, int nblks, int leaky)
{
  int tile = blockIdx.x * 8 + threadIdx.y;
  if (tile >= mblks * nblks) return;
  int mt = tile / nblks, nt = tile % nblks;
  int lane = threadIdx.x;
  int l15  = lane & 15;
  int k0   = (lane & 16) ? 8 : 0;
  const half_t* abase = A  + (size_t)(mt * 32 + l15) * LD + k0;   // row mt*32+l15; +16*LD imm for row1
  const half_t* bbase = Bw + (size_t)(nt * 64 + l15) * LD + k0;   // col block; +c*16*LD imm

  v8f acc[2][4] = {};
  v16h a0 = frag_ld(abase);
  v16h a1 = frag_ld(abase + 16 * LD);
  v16h b[4];
#pragma unroll
  for (int c = 0; c < 4; ++c) b[c] = frag_ld(bbase + c * 16 * LD);

#pragma unroll 2
  for (int k = 32; k < LD; k += 32) {
    // prefetch next chunk while WMMAs below consume the current one
    v16h na0 = frag_ld(abase + k);
    v16h na1 = frag_ld(abase + 16 * LD + k);
    v16h nb[4];
#pragma unroll
    for (int c = 0; c < 4; ++c) nb[c] = frag_ld(bbase + c * 16 * LD + k);
#pragma unroll
    for (int c = 0; c < 4; ++c) {
      acc[0][c] = __builtin_amdgcn_wmma_f32_16x16x32_f16(false, a0, false, b[c], (short)0, acc[0][c], false, false);
      acc[1][c] = __builtin_amdgcn_wmma_f32_16x16x32_f16(false, a1, false, b[c], (short)0, acc[1][c], false, false);
    }
    a0 = na0; a1 = na1;
#pragma unroll
    for (int c = 0; c < 4; ++c) b[c] = nb[c];
  }
#pragma unroll
  for (int c = 0; c < 4; ++c) {
    acc[0][c] = __builtin_amdgcn_wmma_f32_16x16x32_f16(false, a0, false, b[c], (short)0, acc[0][c], false, false);
    acc[1][c] = __builtin_amdgcn_wmma_f32_16x16x32_f16(false, a1, false, b[c], (short)0, acc[1][c], false, false);
  }

  int hi8 = (lane & 16) ? 8 : 0;
#pragma unroll
  for (int c = 0; c < 4; ++c) {
    int n = nt * 64 + c * 16 + l15;
    float bv = (bias && n < nb) ? bias[n] : 0.0f;
#pragma unroll
    for (int mi = 0; mi < 2; ++mi) {
      int mbase = mt * 32 + mi * 16 + hi8;
#pragma unroll
      for (int r = 0; r < 8; ++r) {
        float v = acc[mi][c][r] + bv;
        if (leaky) v = v >= 0.0f ? v : 0.2f * v;
        size_t o = (size_t)(mbase + r) * ldc + n;
        if (C)  C[o]  = v;
        if (Ch) Ch[o] = (half_t)v;
      }
    }
  }
}

// ---------------------------------------------------------------------------
// Implicit-GEMM 3x3 VALID conv, fully templated so every tap/chunk offset is
// an immediate and the flattened (tap x chunk) loop software-pipelines with
// the unroll-2 ping-pong.  One A base address per row, one B base address.
// In: [NSAMP][HIN*HIN][CPAD], W: [9][COUT][CPAD], Out: [NSAMP][P][COUT].
// ---------------------------------------------------------------------------
template<int HIN, int CPAD, int WOUT, int P, int PPAD, int COUT, int CHUNKS>
__global__ void conv_wmma_t(const half_t* __restrict__ In, const half_t* __restrict__ W,
                            const float* __restrict__ bias, half_t* __restrict__ Out)
{
  constexpr int NBLK = COUT / 64;
  constexpr int MBLK = (NSAMP * PPAD) / 32;
  constexpr int NTOT = 9 * CHUNKS;
  int tile = blockIdx.x * 8 + threadIdx.y;
  if (tile >= MBLK * NBLK) return;
  int nt = tile % NBLK;
  int mt = tile / NBLK;
  int lane = threadIdx.x;
  int l15 = lane & 15, hi = (lane >> 4) & 1;
  int k0 = hi * 8;

  // decode the two A rows this lane touches
  int mrow0 = mt * 32 + l15;
  int mrow1 = mrow0 + 16;
  int samp0 = mrow0 / PPAD, p0 = mrow0 % PPAD;
  int samp1 = mrow1 / PPAD, p1 = mrow1 % PPAD;
  bool inr0 = p0 < P, inr1 = p1 < P;
  int y0 = p0 / WOUT, x0 = p0 % WOUT;
  int y1 = p1 / WOUT, x1 = p1 % WOUT;
  const half_t* abase0 = In + ((size_t)samp0 * HIN * HIN + (size_t)y0 * HIN + x0) * CPAD + k0;
  const half_t* abase1 = In + ((size_t)samp1 * HIN * HIN + (size_t)y1 * HIN + x1) * CPAD + k0;
  const half_t* bbase  = W + (size_t)(nt * 64 + l15) * CPAD + k0;   // + (tap*COUT + ci*16)*CPAD imm

  // compile-time offset tables (lane-uniform immediates)
  auto aoff = [](int t) constexpr -> int {
    int tap = t / CHUNKS, c = t % CHUNKS;
    int dy = tap / 3, dx = tap - dy * 3;
    return (dy * HIN + dx) * CPAD + c * 32;
  };
  auto boff = [](int t) constexpr -> int {
    int tap = t / CHUNKS, c = t % CHUNKS;
    return tap * COUT * CPAD + c * 32;
  };

  v8f acc[2][4] = {};
  v16h a0 = {}, a1 = {};
  if (inr0) a0 = frag_ld(abase0 + aoff(0));
  if (inr1) a1 = frag_ld(abase1 + aoff(0));
  v16h b[4];
#pragma unroll
  for (int ci = 0; ci < 4; ++ci) b[ci] = frag_ld(bbase + ci * 16 * CPAD + boff(0));

#pragma unroll 2
  for (int t = 1; t < NTOT; ++t) {
    int ao = aoff(t), bo = boff(t);
    v16h na0 = {}, na1 = {}, nb[4];
    if (inr0) na0 = frag_ld(abase0 + ao);
    if (inr1) na1 = frag_ld(abase1 + ao);
#pragma unroll
    for (int ci = 0; ci < 4; ++ci) nb[ci] = frag_ld(bbase + ci * 16 * CPAD + bo);
#pragma unroll
    for (int ci = 0; ci < 4; ++ci) {
      acc[0][ci] = __builtin_amdgcn_wmma_f32_16x16x32_f16(false, a0, false, b[ci], (short)0, acc[0][ci], false, false);
      acc[1][ci] = __builtin_amdgcn_wmma_f32_16x16x32_f16(false, a1, false, b[ci], (short)0, acc[1][ci], false, false);
    }
    a0 = na0; a1 = na1;
#pragma unroll
    for (int ci = 0; ci < 4; ++ci) b[ci] = nb[ci];
  }
#pragma unroll
  for (int ci = 0; ci < 4; ++ci) {
    acc[0][ci] = __builtin_amdgcn_wmma_f32_16x16x32_f16(false, a0, false, b[ci], (short)0, acc[0][ci], false, false);
    acc[1][ci] = __builtin_amdgcn_wmma_f32_16x16x32_f16(false, a1, false, b[ci], (short)0, acc[1][ci], false, false);
  }

#pragma unroll
  for (int ci = 0; ci < 4; ++ci) {
    int n = nt * 64 + ci * 16 + l15;
    float bv = bias[n];
#pragma unroll
    for (int mi = 0; mi < 2; ++mi) {
#pragma unroll
      for (int r = 0; r < 8; ++r) {
        int m2 = mt * 32 + mi * 16 + hi * 8 + r;
        int s2 = m2 / PPAD, p2 = m2 % PPAD;
        if (p2 < P) {
          float v = acc[mi][ci][r] + bv;
          v = v >= 0.0f ? v : 0.2f * v;
          Out[((size_t)s2 * P + p2) * COUT + n] = (half_t)v;
        }
      }
    }
  }
}

// ------------------------- data movement / prep ----------------------------
__global__ void obs_pad(const float* __restrict__ obs, half_t* __restrict__ act0) {
  size_t i = (size_t)blockIdx.x * 256 + threadIdx.x;
  if (i >= (size_t)NSAMP * 81 * 32) return;
  int c = (int)(i & 31);
  size_t rest = i >> 5;
  int pos = (int)(rest % 81);
  int sample = (int)(rest / 81);
  half_t v = (half_t)0.0f;
  if (c < 6) v = (half_t)obs[((size_t)sample * 6 + c) * 81 + pos];
  act0[i] = v;
}
__global__ void prep_convw(const float* __restrict__ w, half_t* __restrict__ o,
                           int Cout, int Cin, int Cpad) {
  size_t i = (size_t)blockIdx.x * 256 + threadIdx.x;
  if (i >= (size_t)9 * Cout * Cpad) return;
  int c = (int)(i % Cpad);
  size_t t = i / Cpad;
  int n = (int)(t % Cout);
  int tap = (int)(t / Cout);
  o[i] = (c < Cin) ? (half_t)w[((size_t)n * Cin + c) * 9 + tap] : (half_t)0.0f;
}
__global__ void prep_w4(const float* __restrict__ w, half_t* __restrict__ o) {
  size_t i = (size_t)blockIdx.x * 256 + threadIdx.x;
  if (i >= (size_t)256 * 1728) return;
  int k = (int)(i % 1728);
  int n = (int)(i / 1728);
  int tap = k / 192, c = k % 192;
  o[i] = (n < 252) ? (half_t)w[((size_t)n * 192 + c) * 9 + tap] : (half_t)0.0f;
}
__global__ void prep_lin(const float* __restrict__ w, half_t* __restrict__ o,
                         int N, int Kr, int Kp) {
  size_t i = (size_t)blockIdx.x * 256 + threadIdx.x;
  if (i >= (size_t)N * Kp) return;
  int k = (int)(i % Kp);
  int n = (int)(i / Kp);
  o[i] = (k < Kr) ? (half_t)w[(size_t)n * Kr + k] : (half_t)0.0f;
}
__global__ void lat_append(const float* __restrict__ urg, const float* __restrict__ crd,
                           const float* __restrict__ la, half_t* __restrict__ lat) {
  size_t i = (size_t)blockIdx.x * 256 + threadIdx.x;
  if (i >= (size_t)NSAMP * 36) return;
  int col = 252 + (int)(i % 36);
  size_t row = i / 36;
  float v = 0.0f;
  if (col < 254)      v = urg[row * 2 + (col - 252)];
  else if (col < 256) v = crd[row * 2 + (col - 254)];
  else if (col < 261) v = la[row * 5 + (col - 256)];
  lat[row * LATP + col] = (half_t)v;
}
__global__ void init_h(const float* __restrict__ h0, float* __restrict__ H,
                       half_t* __restrict__ Hh) {
  size_t i = (size_t)blockIdx.x * 256 + threadIdx.x;
  if (i >= (size_t)NROWS * HIDN) return;
  float v = h0[i];
  H[i] = v;
  Hh[i] = (half_t)v;
}

// ------------------------- recurrent-step kernels --------------------------
__global__ void ln_kernel(const float* __restrict__ H, const float* __restrict__ g,
                          const float* __restrict__ b, half_t* __restrict__ out) {
  int row = blockIdx.x * 8 + threadIdx.y;   // 1024 rows
  int lane = threadIdx.x;
  const float* hr = H + (size_t)row * HIDN;
  float vals[8];
  float s = 0.0f;
#pragma unroll
  for (int j = 0; j < 8; ++j) { vals[j] = hr[lane * 8 + j]; s += vals[j]; }
  float mean = wave_sum(s) * (1.0f / 256.0f);
  float vs = 0.0f;
#pragma unroll
  for (int j = 0; j < 8; ++j) { float d = vals[j] - mean; vs += d * d; }
  float var = wave_sum(vs) * (1.0f / 256.0f);
  float inv = rsqrtf(var + 1e-5f);
#pragma unroll
  for (int j = 0; j < 8; ++j) {
    int c = lane * 8 + j;
    out[(size_t)row * HIDN + c] = (half_t)((vals[j] - mean) * inv * g[c] + b[c]);
  }
}

// mi[b,i,j] = [ LN(h[b,i]) (256) | pe_w[idx]+pe_b (16) | zeros to 288 ]
__global__ void mi_build(const half_t* __restrict__ hn, const int* __restrict__ rp,
                         const float* __restrict__ pew, const float* __restrict__ peb,
                         half_t* __restrict__ mi, int s) {
  size_t i = (size_t)blockIdx.x * 256 + threadIdx.x;
  if (i >= (size_t)B_DIM * 256 * LATP) return;
  int col = (int)(i % LATP);
  size_t row = i / LATP;                      // b*256 + i*16 + j
  int b = (int)(row >> 8);
  int ii = (int)((row >> 4) & 15);
  int j = (int)(row & 15);
  half_t v;
  if (col < 256) {
    v = hn[(size_t)(b * 16 + ii) * HIDN + col];
  } else if (col < 272) {
    const int* r2 = rp + ((((size_t)b * S_DIM + s) * 16 + ii) * 16 + j) * 2;
    int r0 = r2[0], r1 = r2[1];
    int a0 = r0 < 0 ? -r0 : r0, a1 = r1 < 0 ? -r1 : r1;
    if (a0 > 4 || a1 > 4) { r0 = 0; r1 = 0; }
    int idx = (r0 + 4) + (r1 + 4) * 9;
    int e = col - 256;
    v = (half_t)(pew[idx * 16 + e] + peb[e]);
  } else {
    v = (half_t)0.0f;
  }
  mi[i] = v;
}

// One wave per (b, receiver j, head). lanes0-15: sender scores; all 32: MD sum.
__global__ void attn_kernel(const float* __restrict__ K, const float* __restrict__ V,
                            const float* __restrict__ Q, const unsigned char* __restrict__ cm,
                            int s, int transp, half_t* __restrict__ aggh) {
  int task = blockIdx.x * 8 + threadIdx.y;  // 4096 = 64*16*4
  int lane = threadIdx.x;
  int b = task >> 6, j = (task >> 2) & 15, head = task & 3;
  size_t cmb = ((size_t)b * S_DIM + s) * 2 * 256;
  float sc = -3.402823466e38f, ex = 0.0f, w = 0.0f;
  if (lane < 16) {
    int i = lane;
    const float* kr = K + (size_t)(b * 256 + i * 16 + j) * 128 + head * 32;
    const float* qr = Q + (size_t)(b * 16 + j) * 128 + head * 32;
    float d = 0.0f;
    for (int t = 0; t < 32; ++t) d += kr[t] * qr[t];
    d *= 0.17677669529663687f;  // 1/sqrt(32)
    int ic = transp ? j : i, jc = transp ? i : j;
    unsigned char m = (unsigned char)(cm[cmb + ic * 16 + jc] | cm[cmb + 256 + ic * 16 + jc]);
    sc = m ? d : -3.402823466e38f;
  }
  float mx = wave_max(sc);
  if (lane < 16) ex = __expf(sc - mx);
  float se = wave_sum(ex);
  w = ex / se;
  float acc = 0.0f;
  for (int i = 0; i < 16; ++i) {
    float wi = __shfl(w, i, 32);
    acc += wi * V[(size_t)(b * 256 + i * 16 + j) * 128 + head * 32 + lane];
  }
  aggh[(size_t)(b * 16 + j) * 128 + head * 32 + lane] = (half_t)acc;
}

__device__ __forceinline__ float gru_elem(float ir, float iz, float in_,
                                          float hr, float hz, float hn_, float h) {
  float r = sigmoidf(ir + hr);
  float z = sigmoidf(iz + hz);
  float n = tanhf(in_ + r * hn_);
  return (1.0f - z) * n + z * h;
}

__global__ void gru_combine(const float* __restrict__ GI, const float* __restrict__ GH,
                            float* __restrict__ H, half_t* __restrict__ Hh) {
  size_t idx = (size_t)blockIdx.x * 256 + threadIdx.x;
  if (idx >= (size_t)NROWS * HIDN) return;
  size_t row = idx >> 8;
  int c = (int)(idx & 255);
  size_t g = row * 768 + c;
  float o = gru_elem(GI[g], GI[g + 256], GI[g + 512], GH[g], GH[g + 256], GH[g + 512], H[idx]);
  H[idx] = o;
  Hh[idx] = (half_t)o;
}

__global__ void gru_combine_masked(const float* __restrict__ GI, const float* __restrict__ GH,
                                   float* __restrict__ H, half_t* __restrict__ Hh,
                                   const unsigned char* __restrict__ cm, int s, int transp) {
  __shared__ int um_s;
  int row = blockIdx.x;         // 1024 rows = (b, receiver j)
  int c = threadIdx.x;          // 256 cols
  if (c == 0) {
    int b = row >> 4, j = row & 15;
    size_t base = ((size_t)b * S_DIM + s) * 2 * 256;
    int um = 0;
    for (int i = 0; i < 16; ++i) {
      int ic = transp ? j : i, jc = transp ? i : j;
      um |= cm[base + ic * 16 + jc] | cm[base + 256 + ic * 16 + jc];
    }
    um_s = um;
  }
  __syncthreads();
  size_t idx = (size_t)row * HIDN + c;
  size_t g = (size_t)row * 768 + c;
  float h = H[idx];
  float o = gru_elem(GI[g], GI[g + 256], GI[g + 512], GH[g], GH[g + 256], GH[g + 512], h);
  o = um_s ? o : h;
  H[idx] = o;
  Hh[idx] = (half_t)o;
}

__global__ void buf_copy(const float* __restrict__ H, float* __restrict__ buf, int s) {
  size_t idx = (size_t)blockIdx.x * 256 + threadIdx.x;  // 64*256
  if (idx >= (size_t)B_DIM * HIDN) return;
  int b = (int)(idx >> 8), c = (int)(idx & 255);
  buf[((size_t)s * B_DIM + b) * HIDN + c] = H[(size_t)(b * 16) * HIDN + c];  // agent 0
}

__global__ void head_kernel(const float* __restrict__ buf, const int* __restrict__ steps,
                            const float* __restrict__ advw, const float* __restrict__ advb,
                            const float* __restrict__ stw, const float* __restrict__ stb,
                            float* __restrict__ out) {
  int b = blockIdx.x;
  int lane = threadIdx.x;
  int st = steps[b] - 1;
  const float* h = buf + ((size_t)st * B_DIM + b) * HIDN;
  float hv[8];
#pragma unroll
  for (int j = 0; j < 8; ++j) hv[j] = h[lane * 8 + j];
  float res[6];
  for (int a = 0; a < 6; ++a) {
    const float* w = (a < 5) ? (advw + (size_t)a * HIDN) : stw;
    float s = 0.0f;
#pragma unroll
    for (int j = 0; j < 8; ++j) s += hv[j] * w[lane * 8 + j];
    res[a] = wave_sum(s);
  }
  if (lane == 0) {
    float adv[5], mean = 0.0f;
    for (int a = 0; a < 5; ++a) { adv[a] = res[a] + advb[a]; mean += adv[a]; }
    mean *= 0.2f;
    float stv = res[5] + stb[0];
    for (int a = 0; a < 5; ++a) out[(size_t)b * 5 + a] = stv + adv[a] - mean;
  }
}

// ---------------------------------------------------------------------------
extern "C" void kernel_launch(void* const* d_in, const int* in_sizes, int n_in,
                              void* d_out, int out_size, void* d_ws, size_t ws_size,
                              hipStream_t stream) {
  (void)in_sizes; (void)n_in; (void)out_size; (void)ws_size;
  // ---- inputs (setup_inputs dict order, params flattened depth-first) ----
  const float* obs   = (const float*)d_in[0];
  const float* urg   = (const float*)d_in[1];
  const float* crd   = (const float*)d_in[2];
  const float* lact  = (const float*)d_in[3];
  const int*   steps = (const int*)d_in[4];
  const float* h0    = (const float*)d_in[5];
  const int*   rp    = (const int*)d_in[6];
  const unsigned char* cmask = (const unsigned char*)d_in[7];
  const float* c0w = (const float*)d_in[8],  *c0b = (const float*)d_in[9];
  const float* c1w = (const float*)d_in[10], *c1b = (const float*)d_in[11];
  const float* c2w = (const float*)d_in[12], *c2b = (const float*)d_in[13];
  const float* c3w = (const float*)d_in[14], *c3b = (const float*)d_in[15];
  const float* gwih = (const float*)d_in[16], *gbih = (const float*)d_in[17];
  const float* gwhh = (const float*)d_in[18], *gbhh = (const float*)d_in[19];
  const float* advw = (const float*)d_in[20], *advb = (const float*)d_in[21];
  const float* stw  = (const float*)d_in[22], *stb  = (const float*)d_in[23];
  struct CommIn {
    const float *lng, *lnb, *pew, *peb, *kw, *kb, *vw, *vb, *qw, *qb,
                *agw, *agb, *gwih, *gwhh, *gbih, *gbhh;
  };
  CommIn comm[2];
  for (int m = 0; m < 2; ++m) {
    int base = 24 + m * 16;
    comm[m] = { (const float*)d_in[base+0],  (const float*)d_in[base+1],
                (const float*)d_in[base+2],  (const float*)d_in[base+3],
                (const float*)d_in[base+4],  (const float*)d_in[base+5],
                (const float*)d_in[base+6],  (const float*)d_in[base+7],
                (const float*)d_in[base+8],  (const float*)d_in[base+9],
                (const float*)d_in[base+10], (const float*)d_in[base+11],
                (const float*)d_in[base+12], (const float*)d_in[base+13],
                (const float*)d_in[base+14], (const float*)d_in[base+15] };
  }
  float* out = (float*)d_out;

  // ---- workspace bump allocator ----
  char* ws = (char*)d_ws;
  size_t woff = 0;
  auto alloc = [&](size_t bytes) -> char* {
    char* p = ws + woff;
    woff = (woff + bytes + 255) & ~(size_t)255;
    return p;
  };
  half_t* ACT0 = (half_t*)alloc((size_t)NSAMP * 81 * 32 * 2);
  half_t* ACT1 = (half_t*)alloc((size_t)NSAMP * 49 * 64 * 2);
  half_t* ACT2 = (half_t*)alloc((size_t)NSAMP * 25 * 128 * 2);
  half_t* ACT3 = (half_t*)alloc((size_t)NSAMP * 9 * 192 * 2);
  half_t* LAT  = (half_t*)alloc((size_t)NSAMP * LATP * 2);
  half_t* W1h  = (half_t*)alloc((size_t)9 * 64 * 32 * 2);
  half_t* W2h  = (half_t*)alloc((size_t)9 * 128 * 64 * 2);
  half_t* W3h  = (half_t*)alloc((size_t)9 * 192 * 128 * 2);
  half_t* W4h  = (half_t*)alloc((size_t)256 * 1728 * 2);
  half_t* WIHh = (half_t*)alloc((size_t)768 * LATP * 2);
  half_t* WHHh = (half_t*)alloc((size_t)768 * 256 * 2);
  struct CommW { half_t *kw, *vw, *qw, *agw, *gwih, *gwhh; };
  CommW cw[2];
  for (int m = 0; m < 2; ++m) {
    cw[m].kw   = (half_t*)alloc((size_t)128 * LATP * 2);
    cw[m].vw   = (half_t*)alloc((size_t)128 * LATP * 2);
    cw[m].qw   = (half_t*)alloc((size_t)128 * 256 * 2);
    cw[m].agw  = (half_t*)alloc((size_t)128 * 128 * 2);
    cw[m].gwih = (half_t*)alloc((size_t)768 * 128 * 2);
    cw[m].gwhh = (half_t*)alloc((size_t)768 * 256 * 2);
  }
  half_t* HF16  = (half_t*)alloc((size_t)NROWS * HIDN * 2);
  half_t* HN16  = (half_t*)alloc((size_t)NROWS * HIDN * 2);
  half_t* MI    = (half_t*)alloc((size_t)NSAMP * LATP * 2);   // 16384 x 288
  half_t* AGGH  = (half_t*)alloc((size_t)NROWS * 128 * 2);
  half_t* AGOH  = (half_t*)alloc((size_t)NROWS * 128 * 2);
  float*  H     = (float*)alloc((size_t)NROWS * HIDN * 4);
  float*  GI    = (float*)alloc((size_t)NROWS * 768 * 4);
  float*  GH    = (float*)alloc((size_t)NROWS * 768 * 4);
  float*  KO    = (float*)alloc((size_t)NSAMP * 128 * 4);
  float*  VO    = (float*)alloc((size_t)NSAMP * 128 * 4);
  float*  QO    = (float*)alloc((size_t)NROWS * 128 * 4);
  float*  BUF   = (float*)alloc((size_t)S_DIM * B_DIM * HIDN * 4);

  dim3 blk1(256);
  dim3 blkw(32, 8);
  auto blocks_of = [](size_t total) { return (unsigned)((total + 255) / 256); };
  auto gemm = [&](const half_t* A, const half_t* Bw, int ld,
                  float* C, half_t* Ch, int ldc, const float* bias, int nb,
                  int M, int N, int leaky) {
    int mb = M / 32, nb2 = N / 64;
    dim3 g((mb * nb2 + 7) / 8);
    switch (ld) {
      case 128:
        gemm_f16_t<128><<<g, blkw, 0, stream>>>(A, Bw, C, Ch, ldc, bias, nb, mb, nb2, leaky); break;
      case 256:
        gemm_f16_t<256><<<g, blkw, 0, stream>>>(A, Bw, C, Ch, ldc, bias, nb, mb, nb2, leaky); break;
      case 288:
        gemm_f16_t<288><<<g, blkw, 0, stream>>>(A, Bw, C, Ch, ldc, bias, nb, mb, nb2, leaky); break;
      case 1728:
        gemm_f16_t<1728><<<g, blkw, 0, stream>>>(A, Bw, C, Ch, ldc, bias, nb, mb, nb2, leaky); break;
    }
  };

  // ---- weight prep (f16, GEMM-friendly layouts) ----
  prep_convw<<<blocks_of((size_t)9*64*32),   blk1, 0, stream>>>(c0w, W1h, 64, 6, 32);
  prep_convw<<<blocks_of((size_t)9*128*64),  blk1, 0, stream>>>(c1w, W2h, 128, 64, 64);
  prep_convw<<<blocks_of((size_t)9*192*128), blk1, 0, stream>>>(c2w, W3h, 192, 128, 128);
  prep_w4<<<blocks_of((size_t)256*1728),     blk1, 0, stream>>>(c3w, W4h);
  prep_lin<<<blocks_of((size_t)768*LATP), blk1, 0, stream>>>(gwih, WIHh, 768, 261, LATP);
  prep_lin<<<blocks_of((size_t)768*256),  blk1, 0, stream>>>(gwhh, WHHh, 768, 256, 256);
  for (int m = 0; m < 2; ++m) {
    prep_lin<<<blocks_of((size_t)128*LATP), blk1, 0, stream>>>(comm[m].kw,  cw[m].kw,  128, 272, LATP);
    prep_lin<<<blocks_of((size_t)128*LATP), blk1, 0, stream>>>(comm[m].vw,  cw[m].vw,  128, 272, LATP);
    prep_lin<<<blocks_of((size_t)128*256),  blk1, 0, stream>>>(comm[m].qw,  cw[m].qw,  128, 256, 256);
    prep_lin<<<blocks_of((size_t)128*128),  blk1, 0, stream>>>(comm[m].agw, cw[m].agw, 128, 128, 128);
    prep_lin<<<blocks_of((size_t)768*128),  blk1, 0, stream>>>(comm[m].gwih, cw[m].gwih, 768, 128, 128);
    prep_lin<<<blocks_of((size_t)768*256),  blk1, 0, stream>>>(comm[m].gwhh, cw[m].gwhh, 768, 256, 256);
  }

  // ---- conv stack (implicit-GEMM WMMA, 32x64 tiles/wave, pipelined) ----
  obs_pad<<<blocks_of((size_t)NSAMP*81*32), blk1, 0, stream>>>(obs, ACT0);
  {
    int t1 = ((NSAMP * 64) >> 5) * (64 >> 6);
    conv_wmma_t<9, 32, 7, 49, 64, 64, 1><<<dim3((t1 + 7) / 8), blkw, 0, stream>>>(ACT0, W1h, c0b, ACT1);
    int t2 = ((NSAMP * 32) >> 5) * (128 >> 6);
    conv_wmma_t<7, 64, 5, 25, 32, 128, 2><<<dim3((t2 + 7) / 8), blkw, 0, stream>>>(ACT1, W2h, c1b, ACT2);
    int t3 = ((NSAMP * 16) >> 5) * (192 >> 6);
    conv_wmma_t<5, 128, 3, 9, 16, 192, 4><<<dim3((t3 + 7) / 8), blkw, 0, stream>>>(ACT2, W3h, c2b, ACT3);
  }
  // L4 (1x1 out) == GEMM over NHWC-flattened rows; writes latent cols 0..255
  gemm(ACT3, W4h, 1728, nullptr, LAT, LATP, c3b, 252, NSAMP, 256, 1);
  lat_append<<<blocks_of((size_t)NSAMP*36), blk1, 0, stream>>>(urg, crd, lact, LAT);

  // ---- recurrent loop ----
  init_h<<<blocks_of((size_t)NROWS*HIDN), blk1, 0, stream>>>(h0, H, HF16);
  for (int s = 0; s < S_DIM; ++s) {
    // main GRU
    gemm(LAT + (size_t)s * NROWS * LATP, WIHh, 288, GI, nullptr, 768, gbih, 768, NROWS, 768, 0);
    gemm(HF16, WHHh, 256, GH, nullptr, 768, gbhh, 768, NROWS, 768, 0);
    gru_combine<<<blocks_of((size_t)NROWS*HIDN), blk1, 0, stream>>>(GI, GH, H, HF16);
    // two comm modules: req (transp=0), rep (transp=1)
    for (int m = 0; m < 2; ++m) {
      int tr = m;
      ln_kernel<<<dim3(NROWS / 8), blkw, 0, stream>>>(H, comm[m].lng, comm[m].lnb, HN16);
      mi_build<<<blocks_of((size_t)NSAMP*LATP), blk1, 0, stream>>>(HN16, rp, comm[m].pew, comm[m].peb, MI, s);
      gemm(MI, cw[m].kw, 288, KO, nullptr, 128, comm[m].kb, 128, NSAMP, 128, 0);
      gemm(MI, cw[m].vw, 288, VO, nullptr, 128, comm[m].vb, 128, NSAMP, 128, 0);
      gemm(HN16, cw[m].qw, 256, QO, nullptr, 128, comm[m].qb, 128, NROWS, 128, 0);
      attn_kernel<<<dim3(4096 / 8), blkw, 0, stream>>>(KO, VO, QO, cmask, s, tr, AGGH);
      gemm(AGGH, cw[m].agw, 128, nullptr, AGOH, 128, comm[m].agb, 128, NROWS, 128, 0);
      gemm(AGOH, cw[m].gwih, 128, GI, nullptr, 768, comm[m].gbih, 768, NROWS, 768, 0);
      gemm(HF16, cw[m].gwhh, 256, GH, nullptr, 768, comm[m].gbhh, 768, NROWS, 768, 0);
      gru_combine_masked<<<dim3(NROWS), blk1, 0, stream>>>(GI, GH, H, HF16, cmask, s, tr);
    }
    buf_copy<<<blocks_of((size_t)B_DIM*HIDN), blk1, 0, stream>>>(H, BUF, s);
  }
  // ---- dueling head ----
  head_kernel<<<dim3(B_DIM), dim3(32), 0, stream>>>(BUF, steps, advw, advb, stw, stb, out);
}
